// ArithmeticNps_88785563943773
// MI455X (gfx1250) — matrix operand order repaired and checked by reading, not compiled, not run on hardware
//
#include <hip/hip_runtime.h>
#include <hip/hip_bf16.h>

typedef __attribute__((ext_vector_type(16))) _Float16 v16h;
typedef __attribute__((ext_vector_type(8)))  _Float16 v8h;
typedef __attribute__((ext_vector_type(8)))  float    v8f;

// ---------------------------------------------------------------------------
// CDNA5 16-bit WMMA fragment layouts (wave32):
// A (16x32, MxK): lane L (row = m0 + (L&15)): halves 0..7 = K kb..kb+7,
//                 halves 8..15 = K kb+16..kb+23, kb = k0 + (L>=16 ? 8 : 0).
// B (32x16, KxN): lane L (col = L&15): same K order as A (kb..kb+7, kb+16..kb+23).
// C/D (16x16):    lane L (col = L&15): vgpr v -> row m = v + (L>=16 ? 8 : 0).
// Each wave carries TWO M-tiles (32 batch rows) so every B fragment load
// feeds two v_wmma instructions.
// ---------------------------------------------------------------------------

#define WMMA_F16(a, b, c)                                                     \
    __builtin_amdgcn_wmma_f32_16x16x32_f16(false, (a), false, (b), (short)0,  \
                                           (c), false, false)

__device__ inline void lds_fence() {
    asm volatile("s_wait_dscnt 0" ::: "memory");
}

__device__ inline v16h load_A_lds(const _Float16* p, int lda, int m0, int k0) {
    const int lane = threadIdx.x & 31;
    const int row  = m0 + (lane & 15);
    const int kb   = k0 + ((lane >> 4) << 3);
    const _Float16* q = p + row * lda + kb;
    v8h lo = *(const v8h*)(q);
    v8h hi = *(const v8h*)(q + 16);
    v16h a;
#pragma unroll
    for (int i = 0; i < 8; ++i) { a[i] = lo[i]; a[8 + i] = hi[i]; }
    return a;
}

__device__ inline v16h load_B_gbl(const _Float16* wt /*[N][K] f16*/, int ldk,
                                  int n0, int k0) {
    const int lane = threadIdx.x & 31;
    const int n    = n0 + (lane & 15);
    const int kb   = k0 + ((lane >> 4) << 3);
    const _Float16* q = wt + (size_t)n * ldk + kb;
    v8h lo = *(const v8h*)(q);
    v8h hi = *(const v8h*)(q + 16);
    v16h b;
#pragma unroll
    for (int i = 0; i < 8; ++i) { b[i] = lo[i]; b[8 + i] = hi[i]; }
    return b;
}

__device__ inline v8f bias_frag(const float* bias, int n0) {
    const int lane = threadIdx.x & 31;
    const float bv = bias[n0 + (lane & 15)];
    v8f c;
#pragma unroll
    for (int v = 0; v < 8; ++v) c[v] = bv;
    return c;
}

__device__ inline void store_D(_Float16* out, int ldo, int m0, int n0, v8f d,
                               bool relu) {
    const int lane = threadIdx.x & 31;
    const int n    = n0 + (lane & 15);
    const int mb   = m0 + ((lane >> 4) << 3);
#pragma unroll
    for (int v = 0; v < 8; ++v) {
        float x = d[v];
        if (relu) x = x > 0.f ? x : 0.f;
        out[(mb + v) * ldo + n] = (_Float16)x;
    }
}

__device__ inline void store_D_masked(_Float16* out, int ldo, int m0, int n0,
                                      v8f d, const int* idxr, int r) {
    const int lane = threadIdx.x & 31;
    const int n    = n0 + (lane & 15);
    const int mb   = m0 + ((lane >> 4) << 3);
#pragma unroll
    for (int v = 0; v < 8; ++v) {
        const int m = mb + v;
        if (idxr[m] == r) out[m * ldo + n] = (_Float16)d[v];
    }
}

// D(32xN) = A(32xK from LDS) * B(KxN, weights [N][K] f16) + bias.
// Two M-tiles share every B fragment.
template <bool RELU>
__device__ inline void gemm2(const _Float16* A, int lda, int K,
                             const _Float16* wt, const float* bias, int N,
                             _Float16* out, int ldo) {
    for (int n0 = 0; n0 < N; n0 += 16) {
        v8f acc0 = bias_frag(bias, n0);
        v8f acc1 = acc0;
        for (int k0 = 0; k0 < K; k0 += 32) {
            v16h b  = load_B_gbl(wt, K, n0, k0);
            v16h a0 = load_A_lds(A, lda, 0, k0);
            v16h a1 = load_A_lds(A, lda, 16, k0);
            acc0 = WMMA_F16(a0, b, acc0);
            acc1 = WMMA_F16(a1, b, acc1);
        }
        store_D(out, ldo, 0, n0, acc0, RELU);
        store_D(out, ldo, 16, n0, acc1, RELU);
    }
}

__device__ inline void gemm2_masked(const _Float16* A, int lda, int K,
                                    const _Float16* wt, const float* bias,
                                    int N, _Float16* out, int ldo,
                                    const int* idxr, int r) {
    for (int n0 = 0; n0 < N; n0 += 16) {
        v8f acc0 = bias_frag(bias, n0);
        v8f acc1 = acc0;
        for (int k0 = 0; k0 < K; k0 += 32) {
            v16h b  = load_B_gbl(wt, K, n0, k0);
            v16h a0 = load_A_lds(A, lda, 0, k0);
            v16h a1 = load_A_lds(A, lda, 16, k0);
            acc0 = WMMA_F16(a0, b, acc0);
            acc1 = WMMA_F16(a1, b, acc1);
        }
        store_D_masked(out, ldo, 0, n0, acc0, idxr, r);
        store_D_masked(out, ldo, 16, n0, acc1, idxr, r);
    }
}

// A-fragment for first encoder layer: h1[k]=relu(opv*W1[0,k]+c2*W1[1,k]+b1[k])
__device__ inline v16h make_A_enc(float opv, float c2, const float* W1,
                                  const float* b1, int k0) {
    const int lane = threadIdx.x & 31;
    const int kb   = k0 + ((lane >> 4) << 3);
    v16h a;
#pragma unroll
    for (int i = 0; i < 16; ++i) {
        const int k = kb + ((i < 8) ? i : (8 + i)); // halves 8..15 -> kb+16..23
        float h = opv * W1[k] + c2 * W1[64 + k] + b1[k];
        a[i] = (_Float16)(h > 0.f ? h : 0.f);
    }
    return a;
}

// A-fragment for operator encoder: one-hot row select of er_W1 (3,64).
__device__ inline v16h make_A_er(int opr, const float* W1, const float* b1,
                                 int k0) {
    const int lane = threadIdx.x & 31;
    const int kb   = k0 + ((lane >> 4) << 3);
    v16h a;
#pragma unroll
    for (int i = 0; i < 16; ++i) {
        const int k = kb + ((i < 8) ? i : (8 + i));
        float h = W1[opr * 64 + k] + b1[k];
        a[i] = (_Float16)(h > 0.f ? h : 0.f);
    }
    return a;
}

// ---------------------------------------------------------------------------
// Pre-pass kernels: transpose f32 (K,N) row-major -> f16 [N][K]; read1 GEMM.
// ---------------------------------------------------------------------------
__global__ void transpose_f16(const float* __restrict__ src,
                              _Float16* __restrict__ dst, int K, int N) {
    int i = blockIdx.x * blockDim.x + threadIdx.x;
    if (i >= K * N) return;
    int k = i / N, n = i % N;
    dst[(size_t)n * K + k] = (_Float16)src[(size_t)k * N + n];
}

__global__ void compute_read1(const float* __restrict__ body,
                              const float* __restrict__ Wq,
                              const float* __restrict__ bq,
                              float* __restrict__ out) {
    int n = blockIdx.x;   // 8 rules
    int m = threadIdx.x;  // 32 cols
    float acc = bq[m];
    for (int k = 0; k < 64; ++k) acc += body[n * 64 + k] * Wq[k * 32 + m];
    out[n * 32 + m] = acc;
}

// ---------------------------------------------------------------------------
// Fused main kernel: one wave (32 threads) handles 32 batch elements.
// ---------------------------------------------------------------------------
struct KParams {
    const float* op1;
    const float* op2;
    const int*   opsel;
    const float *eo_W1, *eo_b1, *eo_b2;
    const float *er_W1, *er_b1, *er_b2;
    const float *se_b1, *se_b2;
    const float *s1_bk;
    const float *s2_Wq, *s2_bq, *s2_bk;
    const float *body;
    const float *rh_b1, *rh_b2;
    const float *de_b1, *de_W2, *de_b2;
    const _Float16 *eo2t, *er2t, *se1t, *se2t, *s1kt, *s2kt, *de1t, *rh1t, *rh2t;
    const float* read1;
    float* out;
};

__global__ __launch_bounds__(32) void fused_arith_nps(KParams p) {
    __shared__ _Float16 s_act[32 * 512];      // staging / var_pc
    __shared__ _Float16 s_hidden[3][32 * 256];
    __shared__ _Float16 s_h128[32 * 128];     // intermediate 64/128-wide
    __shared__ _Float16 s_w1[3][32 * 32];     // write1
    __shared__ _Float16 s_k2[2][32 * 16];
    __shared__ _Float16 s_vout[32 * 256];     // var_out
    __shared__ int s_idxr[32], s_idxp[32], s_idxc[32];

    const int base = blockIdx.x * 32;
    const int lane = threadIdx.x & 31;
    const int sub  = lane & 15;  // row within an M-tile

    // per-lane inputs for both M-tiles (tile0 rows 0..15, tile1 rows 16..31)
    const float op1a = p.op1[base + sub];
    const float op1b = p.op1[base + 16 + sub];
    const float op2a = p.op2[base + sub];
    const float op2b = p.op2[base + 16 + sub];
    const int   opra = p.opsel[base + sub];
    const int   oprb = p.opsel[base + 16 + sub];

    // ---- encoders + state-encoder MLP -> hidden[t], t = 0,1,2 -------------
    for (int t = 0; t < 3; ++t) {
        v16h a00, a01, a10, a11;  // [tile][kstep]
        if (t == 0) {
            a00 = make_A_enc(op1a, 0.f, p.eo_W1, p.eo_b1, 0);
            a01 = make_A_enc(op1a, 0.f, p.eo_W1, p.eo_b1, 32);
            a10 = make_A_enc(op1b, 0.f, p.eo_W1, p.eo_b1, 0);
            a11 = make_A_enc(op1b, 0.f, p.eo_W1, p.eo_b1, 32);
        } else if (t == 1) {
            a00 = make_A_enc(op2a, 1.f, p.eo_W1, p.eo_b1, 0);
            a01 = make_A_enc(op2a, 1.f, p.eo_W1, p.eo_b1, 32);
            a10 = make_A_enc(op2b, 1.f, p.eo_W1, p.eo_b1, 0);
            a11 = make_A_enc(op2b, 1.f, p.eo_W1, p.eo_b1, 32);
        } else {
            a00 = make_A_er(opra, p.er_W1, p.er_b1, 0);
            a01 = make_A_er(opra, p.er_W1, p.er_b1, 32);
            a10 = make_A_er(oprb, p.er_W1, p.er_b1, 0);
            a11 = make_A_er(oprb, p.er_W1, p.er_b1, 32);
        }
        const _Float16* w2t = (t < 2) ? p.eo2t : p.er2t;
        const float*    b2  = (t < 2) ? p.eo_b2 : p.er_b2;
        // encoder layer-2: x_e = h1 @ W2 + b2    (32 x 64 x 256)
        for (int n0 = 0; n0 < 256; n0 += 16) {
            v16h b0 = load_B_gbl(w2t, 64, n0, 0);
            v16h b1 = load_B_gbl(w2t, 64, n0, 32);
            v8f acc0 = bias_frag(b2, n0);
            v8f acc1 = acc0;
            acc0 = WMMA_F16(a00, b0, acc0);
            acc0 = WMMA_F16(a01, b1, acc0);
            acc1 = WMMA_F16(a10, b0, acc1);
            acc1 = WMMA_F16(a11, b1, acc1);
            store_D(s_act, 256, 0, n0, acc0, false);
            store_D(s_act, 256, 16, n0, acc1, false);
        }
        lds_fence();
        // state encoder: input = [zeros(256), x_e] -> bottom half of se_W1
        gemm2<true>(s_act, 256, 256, p.se1t, p.se_b1, 64, s_h128, 64);
        lds_fence();
        gemm2<false>(s_h128, 64, 64, p.se2t, p.se_b2, 256, s_hidden[t], 256);
        lds_fence();
    }

    // ---- attention 1: write1 = hidden @ s1_Wk + bk; argmax over 8x3 -------
    for (int t = 0; t < 3; ++t)
        gemm2<false>(s_hidden[t], 256, 256, p.s1kt, p.s1_bk, 32, s_w1[t], 32);
    lds_fence();
    {
        const int row = lane;  // every lane owns one batch element
        float best = -3.0e38f;
        int   bi   = 0;
        for (int n = 0; n < 8; ++n)
            for (int t = 0; t < 3; ++t) {
                float d = 0.f;
                for (int k = 0; k < 32; ++k)
                    d += p.read1[n * 32 + k] * (float)s_w1[t][row * 32 + k];
                if (d > best) { best = d; bi = n * 3 + t; }  // scale monotonic
            }
        s_idxr[row] = bi / 3;
        s_idxp[row] = bi % 3;
    }
    lds_fence();

    // ---- attention 2: k2 = hid2 @ s2_Wk + bk; q2 per-lane; argmax over 2 --
    for (int t = 0; t < 2; ++t)
        gemm2<false>(s_hidden[t], 256, 256, p.s2kt, p.s2_bk, 16, s_k2[t], 16);
    lds_fence();
    {
        const int    row   = lane;
        const int    ir    = s_idxr[row];
        const float* bodyr = p.body + ir * 64;
        float a0 = 0.f, a1 = 0.f;
        for (int m = 0; m < 16; ++m) {
            float q = p.s2_bq[m];
            for (int k = 0; k < 64; ++k) q += bodyr[k] * p.s2_Wq[k * 16 + m];
            a0 += q * (float)s_k2[0][row * 16 + m];
            a1 += q * (float)s_k2[1][row * 16 + m];
        }
        s_idxc[row] = (a1 > a0) ? 1 : 0;
    }
    lds_fence();

    // ---- build var_pc = [hidden[idx_p], hidden[idx_c]] in s_act (32x512) --
    {
        const int row = lane;
        const _Float16* sp = &s_hidden[s_idxp[row]][row * 256];
        const _Float16* sc = &s_hidden[s_idxc[row]][row * 256];
        _Float16* d = &s_act[row * 512];
        for (int i = 0; i < 256; i += 8) {
            *(v8h*)(d + i)       = *(const v8h*)(sp + i);
            *(v8h*)(d + 256 + i) = *(const v8h*)(sc + i);
        }
    }
    lds_fence();

    // ---- gathered-rule MLP: only rules present in this 32-tile -----------
    for (int r = 0; r < 8; ++r) {
        const bool mine = (s_idxr[lane] == r);
        unsigned long long m = __ballot((int)mine);
        if ((m & 0xffffffffull) == 0ull) continue;  // wave-uniform skip
        if (r + 1 < 8)
            __builtin_prefetch(p.rh1t + (size_t)(r + 1) * 128 * 512, 0, 1);
        gemm2<true>(s_act, 512, 512, p.rh1t + (size_t)r * 128 * 512,
                    p.rh_b1 + r * 128, 128, s_h128, 128);
        lds_fence();
        gemm2_masked(s_h128, 128, 128, p.rh2t + (size_t)r * 256 * 128,
                     p.rh_b2 + r * 256, 256, s_vout, 256, s_idxr, r);
        lds_fence();
    }

    // ---- decoder MLP: 256 -> 64 (relu) -> 1 ------------------------------
    gemm2<true>(s_vout, 256, 256, p.de1t, p.de_b1, 64, s_h128, 64);
    lds_fence();
    {
        const int row = lane;
        float acc = p.de_b2[0];
        for (int k = 0; k < 64; ++k)
            acc += (float)s_h128[row * 64 + k] * p.de_W2[k];
        p.out[base + row] = acc;
    }
}

// ---------------------------------------------------------------------------
extern "C" void kernel_launch(void* const* d_in, const int* in_sizes, int n_in,
                              void* d_out, int out_size, void* d_ws,
                              size_t ws_size, hipStream_t stream) {
    const float* op1   = (const float*)d_in[0];
    const float* op2   = (const float*)d_in[1];
    const int*   opsel = (const int*)d_in[2];
    const float* eo_W1 = (const float*)d_in[3];
    const float* eo_b1 = (const float*)d_in[4];
    const float* eo_W2 = (const float*)d_in[5];
    const float* eo_b2 = (const float*)d_in[6];
    const float* er_W1 = (const float*)d_in[7];
    const float* er_b1 = (const float*)d_in[8];
    const float* er_W2 = (const float*)d_in[9];
    const float* er_b2 = (const float*)d_in[10];
    const float* de_W1 = (const float*)d_in[11];
    const float* de_b1 = (const float*)d_in[12];
    const float* de_W2 = (const float*)d_in[13];
    const float* de_b2 = (const float*)d_in[14];
    const float* se_W1 = (const float*)d_in[15];
    const float* se_b1 = (const float*)d_in[16];
    const float* se_W2 = (const float*)d_in[17];
    const float* se_b2 = (const float*)d_in[18];
    const float* rh_W1 = (const float*)d_in[19];
    const float* rh_b1 = (const float*)d_in[20];
    const float* rh_W2 = (const float*)d_in[21];
    const float* rh_b2 = (const float*)d_in[22];
    const float* body  = (const float*)d_in[23];
    const float* s1_Wq = (const float*)d_in[24];
    const float* s1_bq = (const float*)d_in[25];
    const float* s1_Wk = (const float*)d_in[26];
    const float* s1_bk = (const float*)d_in[27];
    const float* s2_Wq = (const float*)d_in[28];
    const float* s2_bq = (const float*)d_in[29];
    const float* s2_Wk = (const float*)d_in[30];
    const float* s2_bk = (const float*)d_in[31];

    const int B = in_sizes[0];

    // f16 weight region in workspace, [N][K] layouts for WMMA B-fragments
    _Float16* H = (_Float16*)d_ws;
    size_t o = 0;
    auto halfbuf = [&](size_t n) { _Float16* q = H + o; o += n; return q; };
    _Float16* eo2t = halfbuf(256 * 64);
    _Float16* er2t = halfbuf(256 * 64);
    _Float16* se1t = halfbuf(64 * 256);
    _Float16* se2t = halfbuf(256 * 64);
    _Float16* s1kt = halfbuf(32 * 256);
    _Float16* s2kt = halfbuf(16 * 256);
    _Float16* de1t = halfbuf(64 * 256);
    _Float16* rh1t = halfbuf((size_t)8 * 128 * 512);
    _Float16* rh2t = halfbuf((size_t)8 * 256 * 128);
    float* read1 = (float*)(H + o);  // 8*32 floats, offset is 16B aligned

    auto T = [&](const float* src, _Float16* dst, int K, int N) {
        int tot = K * N;
        transpose_f16<<<(tot + 255) / 256, 256, 0, stream>>>(src, dst, K, N);
    };
    T(eo_W2, eo2t, 64, 256);
    T(er_W2, er2t, 64, 256);
    T(se_W1 + 256 * 64, se1t, 256, 64);  // only bottom half (state is zero)
    T(se_W2, se2t, 64, 256);
    T(s1_Wk, s1kt, 256, 32);
    T(s2_Wk, s2kt, 256, 16);
    T(de_W1, de1t, 256, 64);
    for (int r = 0; r < 8; ++r) {
        T(rh_W1 + (size_t)r * 512 * 128, rh1t + (size_t)r * 128 * 512, 512, 128);
        T(rh_W2 + (size_t)r * 128 * 256, rh2t + (size_t)r * 256 * 128, 128, 256);
    }
    compute_read1<<<8, 32, 0, stream>>>(body, s1_Wq, s1_bq, read1);

    KParams p;
    p.op1 = op1; p.op2 = op2; p.opsel = opsel;
    p.eo_W1 = eo_W1; p.eo_b1 = eo_b1; p.eo_b2 = eo_b2;
    p.er_W1 = er_W1; p.er_b1 = er_b1; p.er_b2 = er_b2;
    p.se_b1 = se_b1; p.se_b2 = se_b2;
    p.s1_bk = s1_bk;
    p.s2_Wq = s2_Wq; p.s2_bq = s2_bq; p.s2_bk = s2_bk;
    p.body = body;
    p.rh_b1 = rh_b1; p.rh_b2 = rh_b2;
    p.de_b1 = de_b1; p.de_W2 = de_W2; p.de_b2 = de_b2;
    p.eo2t = eo2t; p.er2t = er2t; p.se1t = se1t; p.se2t = se2t;
    p.s1kt = s1kt; p.s2kt = s2kt; p.de1t = de1t;
    p.rh1t = rh1t; p.rh2t = rh2t;
    p.read1 = read1;
    p.out = (float*)d_out;

    fused_arith_nps<<<dim3(B / 32), dim3(32), 0, stream>>>(p);
}
